// ArcFaceLoss_44126493999503
// MI455X (gfx1250) — matrix operand order repaired and checked by reading, not compiled, not run on hardware
//
#include <hip/hip_runtime.h>
#include <stdint.h>

// ---------------- problem constants ----------------
#define IN_F   512
#define BATCH  512
#define NCLS   100000
#define NTILE  6250          // NCLS / 16

#define COS_M  0.8775825618903728f
#define SIN_M  0.4794255386042030f
#define THR    (-0.8775825618903728f)
#define MMV    0.2397127693021015f
#define SCALE  64.0f

typedef __attribute__((ext_vector_type(16))) __bf16 v16bf;
typedef __attribute__((ext_vector_type(8)))  float  v8f;

// v16bf member gives this union 32-byte alignment (>= uint4's 16B requirement)
union BPack { uint32_t u[8]; uint4 q[2]; v16bf v; };

__device__ __forceinline__ uint32_t fbits(float a) { return __builtin_bit_cast(uint32_t, a); }
__device__ __forceinline__ float    bhi_f(float a) { return __builtin_bit_cast(float, fbits(a) & 0xffff0000u); }
// pack two bf16 (truncated) values, a -> [15:0], b -> [31:16]
__device__ __forceinline__ uint32_t pk_bf16(float a, float b) {
    return (fbits(a) >> 16) | (fbits(b) & 0xffff0000u);
}

// ---------------- kernel 1: normalize x rows, emit bf16 hi/lo ----------------
__global__ void __launch_bounds__(128) prep_x(const float* __restrict__ x,
                                              uint16_t* __restrict__ xhi,
                                              uint16_t* __restrict__ xlo) {
    const int row = blockIdx.x;
    const int t   = threadIdx.x;          // 128 threads
    const float* xr = x + (size_t)row * IN_F;
    float vals[4];
    float ssq = 0.f;
#pragma unroll
    for (int i = 0; i < 4; ++i) { vals[i] = xr[t + i * 128]; ssq += vals[i] * vals[i]; }
    __shared__ float red[128];
    red[t] = ssq;
    __syncthreads();
    for (int s = 64; s > 0; s >>= 1) {
        if (t < s) red[t] += red[t + s];
        __syncthreads();
    }
    const float inv = 1.0f / fmaxf(sqrtf(red[0]), 1e-12f);
#pragma unroll
    for (int i = 0; i < 4; ++i) {
        const float xn = vals[i] * inv;
        const float h  = bhi_f(xn);
        const float l  = xn - h;
        const int idx  = row * IN_F + t + i * 128;
        xhi[idx] = (uint16_t)(fbits(xn) >> 16);
        xlo[idx] = (uint16_t)(fbits(l)  >> 16);
    }
}

// ---------------- kernel 2: fused WMMA GEMM + margin + per-tile softmax stats ----
// grid: 6250 blocks (one 16-class tile each), 256 threads = 8 waves.
// wave w handles rows [w*64, w*64+64) as 4 row-tiles of 16.
__global__ void __launch_bounds__(256) arcface_gemm(const float* __restrict__ w,
                                                    const int*   __restrict__ y,
                                                    const uint16_t* __restrict__ xhi,
                                                    const uint16_t* __restrict__ xlo,
                                                    float* __restrict__ pm,
                                                    float* __restrict__ ps,
                                                    float* __restrict__ tgt) {
    const int ct    = blockIdx.x;
    const int cbase = ct * 16;
    const int lane  = threadIdx.x & 31;
    const int wv    = threadIdx.x >> 5;
    const int lg    = lane >> 4;        // 16-lane group (0/1)
    const int ln    = lane & 15;
    const int c     = cbase + ln;       // this lane's class column
    const int khB   = lg * 16;          // B k-half within 32-chunk
    const int khA   = lg * 16;          // A k-half within 32-chunk
    const int rb    = wv * 64;          // wave row base

    v8f acc[4];
#pragma unroll
    for (int t = 0; t < 4; ++t)
#pragma unroll
        for (int v = 0; v < 8; ++v) acc[t][v] = 0.f;

    float ssq = 0.f;
    const float* wrow = w + (size_t)c * IN_F;

    for (int kc = 0; kc < IN_F; kc += 32) {
        // ---- load 16 consecutive f32 of this class row; split to bf16 hi/lo ----
        const float4* wp = (const float4*)(wrow + kc + khB);
        float f[16];
        {
            float4 q0 = wp[0], q1 = wp[1], q2 = wp[2], q3 = wp[3];
            f[0]=q0.x; f[1]=q0.y; f[2]=q0.z; f[3]=q0.w;
            f[4]=q1.x; f[5]=q1.y; f[6]=q1.z; f[7]=q1.w;
            f[8]=q2.x; f[9]=q2.y; f[10]=q2.z; f[11]=q2.w;
            f[12]=q3.x; f[13]=q3.y; f[14]=q3.z; f[15]=q3.w;
        }
        BPack bhi, blo;
#pragma unroll
        for (int j = 0; j < 8; ++j) {
            const float a = f[2 * j], b = f[2 * j + 1];
            ssq += a * a + b * b;
            const float ah = bhi_f(a), bh = bhi_f(b);
            bhi.u[j] = pk_bf16(a, b);
            blo.u[j] = pk_bf16(a - ah, b - bh);
        }
        // ---- 4 row-tiles against this B chunk ----
#pragma unroll
        for (int t = 0; t < 4; ++t) {
            const int r = rb + t * 16 + ln;
            const uint16_t* hp = xhi + (size_t)r * IN_F + kc;
            const uint16_t* lp = xlo + (size_t)r * IN_F + kc;
            BPack ah, al;
            ah.q[0] = *(const uint4*)(hp + khA);
            ah.q[1] = *(const uint4*)(hp + khA + 8);
            al.q[0] = *(const uint4*)(lp + khA);
            al.q[1] = *(const uint4*)(lp + khA + 8);
            acc[t] = __builtin_amdgcn_wmma_f32_16x16x32_bf16(false, ah.v, false, bhi.v,
                                                             (short)0, acc[t], false, false);
            acc[t] = __builtin_amdgcn_wmma_f32_16x16x32_bf16(false, ah.v, false, blo.v,
                                                             (short)0, acc[t], false, false);
            acc[t] = __builtin_amdgcn_wmma_f32_16x16x32_bf16(false, al.v, false, bhi.v,
                                                             (short)0, acc[t], false, false);
        }
    }

    // weight-row inverse norm: combine the two k-halves (lanes l and l^16)
    ssq += __shfl_xor(ssq, 16, 32);
    const float invnw = 1.0f / fmaxf(sqrtf(ssq), 1e-12f);

    // ---- epilogue: margin + per-row (max, sumexp) over this 16-class tile ----
#pragma unroll
    for (int t = 0; t < 4; ++t) {
        float lgv[8], mx[8], se[8];
#pragma unroll
        for (int v = 0; v < 8; ++v) {
            const int m  = rb + t * 16 + v + lg * 8;   // C layout: M=v (+8 for hi group)
            const float cs = acc[t][v] * invnw;
            float logit = SCALE * cs;
            if (y[m] == c) {
                const float sn  = sqrtf(fmaxf(1.f - cs * cs, 0.f));
                const float cwm = cs * COS_M - sn * SIN_M;
                const float md  = (cs > THR) ? cwm : (cs - MMV);
                logit = SCALE * md;
                tgt[m] = logit;            // exactly one lane grid-wide per row
            }
            lgv[v] = logit;
        }
#pragma unroll
        for (int v = 0; v < 8; ++v) {
            float mv = lgv[v];
            for (int msk = 1; msk < 16; msk <<= 1)
                mv = fmaxf(mv, __shfl_xor(mv, msk, 32));
            float e = __expf(lgv[v] - mv);
            for (int msk = 1; msk < 16; msk <<= 1)
                e += __shfl_xor(e, msk, 32);
            mx[v] = mv; se[v] = e;
        }
        if (ln == 0) {                      // lanes 0 and 16 write 8 rows each
#pragma unroll
            for (int v = 0; v < 8; ++v) {
                const int m = rb + t * 16 + v + lg * 8;
                pm[(size_t)m * NTILE + ct] = mx[v];
                ps[(size_t)m * NTILE + ct] = se[v];
            }
        }
    }
}

// ---------------- kernel 3: per-row logsumexp merge over 6250 tiles -----------
__global__ void __launch_bounds__(32) merge_rows(const float* __restrict__ pm,
                                                 const float* __restrict__ ps,
                                                 float* __restrict__ lse) {
    const int r = blockIdx.x;
    const int l = threadIdx.x;          // one wave
    float M = -3.0e38f, S = 0.f;
    for (int t = l; t < NTILE; t += 32) {
        const float m2 = pm[(size_t)r * NTILE + t];
        const float s2 = ps[(size_t)r * NTILE + t];
        const float nm = fmaxf(M, m2);
        S = S * __expf(M - nm) + s2 * __expf(m2 - nm);
        M = nm;
    }
    for (int msk = 1; msk < 32; msk <<= 1) {
        const float m2 = __shfl_xor(M, msk, 32);
        const float s2 = __shfl_xor(S, msk, 32);
        const float nm = fmaxf(M, m2);
        S = S * __expf(M - nm) + s2 * __expf(m2 - nm);
        M = nm;
    }
    if (l == 0) lse[r] = M + __logf(S);
}

// ---------------- kernel 4: mean NLL ----------------
__global__ void __launch_bounds__(512) final_loss(const float* __restrict__ lse,
                                                  const float* __restrict__ tgt,
                                                  float* __restrict__ out) {
    __shared__ float red[512];
    const int t = threadIdx.x;
    red[t] = lse[t] - tgt[t];
    __syncthreads();
    for (int s = 256; s > 0; s >>= 1) {
        if (t < s) red[t] += red[t + s];
        __syncthreads();
    }
    if (t == 0) out[0] = red[0] * (1.0f / (float)BATCH);
}

// ---------------- host entry ----------------
extern "C" void kernel_launch(void* const* d_in, const int* in_sizes, int n_in,
                              void* d_out, int out_size, void* d_ws, size_t ws_size,
                              hipStream_t stream) {
    (void)in_sizes; (void)n_in; (void)out_size; (void)ws_size;
    const float* x = (const float*)d_in[0];
    const int*   y = (const int*)d_in[1];
    const float* w = (const float*)d_in[2];
    float* out = (float*)d_out;

    // workspace layout
    uint16_t* xhi = (uint16_t*)d_ws;                       // 512*512 bf16
    uint16_t* xlo = xhi + (size_t)BATCH * IN_F;            // 512*512 bf16
    float*    tgt = (float*)(xlo + (size_t)BATCH * IN_F);  // 512 f32
    float*    lse = tgt + BATCH;                           // 512 f32
    float*    pm  = lse + BATCH;                           // 512*6250 f32
    float*    ps  = pm + (size_t)BATCH * NTILE;            // 512*6250 f32

    prep_x<<<BATCH, 128, 0, stream>>>(x, xhi, xlo);
    arcface_gemm<<<NTILE, 256, 0, stream>>>(w, y, xhi, xlo, pm, ps, tgt);
    merge_rows<<<BATCH, 32, 0, stream>>>(pm, ps, lse);
    final_loss<<<1, 512, 0, stream>>>(lse, tgt, out);
}
